// SingleStreamAttention_37065567765178
// MI455X (gfx1250) — matrix operand order repaired and checked by reading, not compiled, not run on hardware
//
#include <hip/hip_runtime.h>
#include <hip/hip_bf16.h>
#include <math.h>

typedef __attribute__((ext_vector_type(2)))  float   v2f;
typedef __attribute__((ext_vector_type(8)))  float   v8f;
typedef __attribute__((ext_vector_type(16))) __bf16  v16bf;
typedef __attribute__((ext_vector_type(8)))  __bf16  v8bf;
typedef __attribute__((ext_vector_type(4)))  __bf16  v4bf;

#define RMS_EPS      1e-6f
#define SCALE_EPS_F  1e-8f
#define LOG_ROPE     9.210340371976184f   /* ln(10000) */

#define BM 128
#define BN 128
#define BK 32
#define LDA_P  36   /* f32 LDS row stride for the f32 (KV) GEMM            */
#define LDB_P  132
#define LDSTR  40   /* bf16 LDS row stride: 80B rows, b128-aligned frags   */

// ---------------------------------------------------------------------------
// 1) min/max of the two attention maps (single workgroup, N = 21504)
// ---------------------------------------------------------------------------
__global__ __launch_bounds__(256) void minmax_kernel(const float* __restrict__ map,
                                                     int N, float* __restrict__ out4) {
  int t = threadIdx.x;
  float mn0 = __builtin_inff(), mx0 = -__builtin_inff();
  float mn1 = __builtin_inff(), mx1 = -__builtin_inff();
  for (int i = t; i < N; i += 256) {
    float a = map[i], b = map[N + i];
    mn0 = fminf(mn0, a); mx0 = fmaxf(mx0, a);
    mn1 = fminf(mn1, b); mx1 = fmaxf(mx1, b);
  }
  for (int off = 16; off; off >>= 1) {
    mn0 = fminf(mn0, __shfl_xor(mn0, off, 32));
    mx0 = fmaxf(mx0, __shfl_xor(mx0, off, 32));
    mn1 = fminf(mn1, __shfl_xor(mn1, off, 32));
    mx1 = fmaxf(mx1, __shfl_xor(mx1, off, 32));
  }
  __shared__ float s[4][8];
  int w = t >> 5;
  if ((t & 31) == 0) { s[0][w]=mn0; s[1][w]=mx0; s[2][w]=mn1; s[3][w]=mx1; }
  __syncthreads();
  if (t == 0) {
    float a=s[0][0], b=s[1][0], c=s[2][0], d=s[3][0];
    for (int i=1;i<8;i++){ a=fminf(a,s[0][i]); b=fmaxf(b,s[1][i]);
                           c=fminf(c,s[2][i]); d=fmaxf(d,s[3][i]); }
    out4[0]=a; out4[1]=b; out4[2]=c; out4[3]=d;
  }
}

// ---------------------------------------------------------------------------
// 2) per-token rope position
// ---------------------------------------------------------------------------
__global__ __launch_bounds__(256) void pos_kernel(const float* __restrict__ map, int N,
                                                  const float* __restrict__ mm,
                                                  float* __restrict__ pos) {
  int i = blockIdx.x * 256 + threadIdx.x;
  if (i >= N) return;
  float m0 = map[i], m1 = map[N + i];
  float h1 = (m0 - mm[0]) / (mm[1] - mm[0] + SCALE_EPS_F) * 4.0f;
  float h2 = (m1 - mm[2]) / (mm[3] - mm[2] + SCALE_EPS_F) * 4.0f + 20.0f;
  pos[i] = (m0 >= m1) ? h1 : h2;
}

// ---------------------------------------------------------------------------
// bf16 split helper (RNE via hardware cvt), returns by value
// ---------------------------------------------------------------------------
struct bf2 { __bf16 h, l; };
__device__ __forceinline__ bf2 split_bf16(float x) {
  bf2 r;
  r.h = (__bf16)x;
  r.l = (__bf16)(x - (float)r.h);
  return r;
}

// ---------------------------------------------------------------------------
// 3) weight transpose + bf16 hi/lo split:  W (KxN f32) -> Thi/Tlo (NxK bf16)
// ---------------------------------------------------------------------------
__global__ __launch_bounds__(256) void transpose_split_kernel(
    const float* __restrict__ W, __bf16* __restrict__ Thi, __bf16* __restrict__ Tlo,
    int K, int N) {
  __shared__ float tile[32][33];
  const int k0 = blockIdx.y * 32, n0 = blockIdx.x * 32;
  const int t = threadIdx.x;
  { int r = t >> 3, c = (t & 7) * 4;
    float4 v = *(const float4*)(W + (size_t)(k0 + r) * N + n0 + c);
    tile[r][c] = v.x; tile[r][c+1] = v.y; tile[r][c+2] = v.z; tile[r][c+3] = v.w; }
  __syncthreads();
  int n = t >> 3, kg = (t & 7) * 4;
  v4bf hv, lv;
  #pragma unroll
  for (int i = 0; i < 4; i++) {
    bf2 s = split_bf16(tile[kg + i][n]);
    hv[i] = s.h; lv[i] = s.l;
  }
  size_t o = (size_t)(n0 + n) * K + k0 + kg;
  *(v4bf*)(Thi + o) = hv;
  *(v4bf*)(Tlo + o) = lv;
}

// ---------------------------------------------------------------------------
// shared register epilogue: bias (+ optional rms_norm + rope) + store
// MODE 0: Q (rms+rope, pos from pos_buf). MODE 1: KV (k half rms+rope 2/22).
// MODE 2: plain bias store.
// acc layout: tile nt, VGPR v, lane l -> row wave*16 + v + 8*(l>>4), col nt*16+(l&15)
// ---------------------------------------------------------------------------
template<int MODE>
__device__ __forceinline__ void gemm_epilogue(
    v8f acc[8], const float* __restrict__ bias, float* __restrict__ C,
    int M, int N, int m0, int n0, int wave, int lane,
    const float* __restrict__ pos_buf, const float* __restrict__ norm_w) {
  const int lhalf = lane >> 4, lmod = lane & 15;
  const int rbase = wave * 16 + 8 * lhalf;
  float bval[8];
  #pragma unroll
  for (int nt = 0; nt < 8; nt++) bval[nt] = bias[n0 + nt * 16 + lmod];

  const bool plain = (MODE == 2) || (MODE == 1 && n0 >= (N >> 1));
  if (plain) {
    #pragma unroll
    for (int v = 0; v < 8; v++) {
      int rg = m0 + rbase + v;
      if (rg < M) {
        #pragma unroll
        for (int nt = 0; nt < 8; nt++)
          C[(size_t)rg * N + n0 + nt * 16 + lmod] = acc[nt][v] + bval[nt];
      }
    }
  } else {
    float nw[8], invf[8];
    #pragma unroll
    for (int nt = 0; nt < 8; nt++) {
      int d = nt * 16 + lmod;
      nw[nt]   = norm_w[d];
      invf[nt] = __expf(-(float)(d & ~1) * (LOG_ROPE / 128.0f));
    }
    #pragma unroll
    for (int v = 0; v < 8; v++) {
      int rg = m0 + rbase + v;
      float vals[8], ss = 0.f;
      #pragma unroll
      for (int nt = 0; nt < 8; nt++) { float x = acc[nt][v] + bval[nt]; vals[nt] = x; ss += x * x; }
      ss += __shfl_xor(ss, 1, 32);
      ss += __shfl_xor(ss, 2, 32);
      ss += __shfl_xor(ss, 4, 32);
      ss += __shfl_xor(ss, 8, 32);
      float scale = rsqrtf(ss * (1.0f / 128.0f) + RMS_EPS);
      float p;
      if (MODE == 0) p = pos_buf[rg < M ? rg : 0];
      else           p = ((rg & 31) < 16) ? 2.0f : 22.0f;
      #pragma unroll
      for (int nt = 0; nt < 8; nt++) {
        int d = nt * 16 + lmod;
        float xn = vals[nt] * scale * nw[nt];
        float other = __shfl_xor(xn, 1, 32);
        float th = p * invf[nt];
        float cs = __cosf(th), sn = __sinf(th);
        float o = (d & 1) ? (xn * cs + other * sn) : (xn * cs - other * sn);
        if (rg < M) C[(size_t)rg * N + n0 + d] = o;
      }
    }
  }
}

// ---------------------------------------------------------------------------
// 4) bf16x3 split-precision WMMA GEMM (the two big 21504x3072x3072 GEMMs).
//    C = A @ W + bias with A split to bf16 hi/lo during the LDS fill and W
//    pre-split/transposed (NxK bf16).  A.B ~= Ahi.Bhi + Ahi.Blo + Alo.Bhi,
//    fp32 accumulation -> ~2^-16 rel error, 3 WMMAs per K=32 (vs 8 for f32).
// ---------------------------------------------------------------------------
template<int MODE>
__global__ __launch_bounds__(256) void gemm_bf16x3_kernel(
    const float* __restrict__ A, const __bf16* __restrict__ BtHi,
    const __bf16* __restrict__ BtLo, const float* __restrict__ bias,
    float* __restrict__ C, int M, int N, int K,
    const float* __restrict__ pos_buf, const float* __restrict__ norm_w)
{
  __shared__ __bf16 lAhi[BM * LDSTR];
  __shared__ __bf16 lAlo[BM * LDSTR];
  __shared__ __bf16 lBhi[BN * LDSTR];
  __shared__ __bf16 lBlo[BN * LDSTR];

  const int tid  = threadIdx.x;
  const int wave = tid >> 5;
  const int lane = tid & 31;
  const int lhalf = lane >> 4, lmod = lane & 15;
  const int n0 = blockIdx.x * BN;
  const int m0 = blockIdx.y * BM;

  v8f acc[8] = {};

  const int ar = tid >> 1,   ac = (tid & 1) * 16;   // A fill: 128 rows x 32 k (f32->bf16x2)
  const int bcol = tid >> 1, bk = (tid & 1) * 16;   // B fill: 128 cols x 32 k (bf16 copy)

  for (int kb = 0; kb < K; kb += BK) {
    // ---- A: global f32 -> split -> LDS bf16 planes ----
    {
      int grow = m0 + ar;
      #pragma unroll
      for (int i = 0; i < 4; i++) {
        float4 v = (grow < M)
          ? *(const float4*)(A + (size_t)grow * K + kb + ac + 4 * i)
          : make_float4(0.f, 0.f, 0.f, 0.f);
        v4bf hv, lv;
        bf2 s0 = split_bf16(v.x); hv[0] = s0.h; lv[0] = s0.l;
        bf2 s1 = split_bf16(v.y); hv[1] = s1.h; lv[1] = s1.l;
        bf2 s2 = split_bf16(v.z); hv[2] = s2.h; lv[2] = s2.l;
        bf2 s3 = split_bf16(v.w); hv[3] = s3.h; lv[3] = s3.l;
        *(v4bf*)(lAhi + ar * LDSTR + ac + 4 * i) = hv;
        *(v4bf*)(lAlo + ar * LDSTR + ac + 4 * i) = lv;
      }
    }
    // ---- B: global bf16 (NxK) -> LDS, K-contiguous rows ----
    {
      size_t o = (size_t)(n0 + bcol) * K + kb + bk;
      *(v8bf*)(lBhi + bcol * LDSTR + bk)     = *(const v8bf*)(BtHi + o);
      *(v8bf*)(lBhi + bcol * LDSTR + bk + 8) = *(const v8bf*)(BtHi + o + 8);
      *(v8bf*)(lBlo + bcol * LDSTR + bk)     = *(const v8bf*)(BtLo + o);
      *(v8bf*)(lBlo + bcol * LDSTR + bk + 8) = *(const v8bf*)(BtLo + o + 8);
    }
    __syncthreads();

    // ---- fragments + 3 WMMAs per N-tile ----
    const int arow = wave * 16 + lmod;
    v16bf ahi, alo;
    ((v8bf*)&ahi)[0] = *(const v8bf*)(lAhi + arow * LDSTR + 8 * lhalf);
    ((v8bf*)&ahi)[1] = *(const v8bf*)(lAhi + arow * LDSTR + 16 + 8 * lhalf);
    ((v8bf*)&alo)[0] = *(const v8bf*)(lAlo + arow * LDSTR + 8 * lhalf);
    ((v8bf*)&alo)[1] = *(const v8bf*)(lAlo + arow * LDSTR + 16 + 8 * lhalf);
    #pragma unroll
    for (int nt = 0; nt < 8; nt++) {
      const int col = nt * 16 + lmod;
      v16bf bhi, blo;
      ((v8bf*)&bhi)[0] = *(const v8bf*)(lBhi + col * LDSTR + 8 * lhalf);
      ((v8bf*)&bhi)[1] = *(const v8bf*)(lBhi + col * LDSTR + 16 + 8 * lhalf);
      ((v8bf*)&blo)[0] = *(const v8bf*)(lBlo + col * LDSTR + 8 * lhalf);
      ((v8bf*)&blo)[1] = *(const v8bf*)(lBlo + col * LDSTR + 16 + 8 * lhalf);
      acc[nt] = __builtin_amdgcn_wmma_f32_16x16x32_bf16(
          false, alo, false, bhi, (short)0, acc[nt], false, false);
      acc[nt] = __builtin_amdgcn_wmma_f32_16x16x32_bf16(
          false, ahi, false, blo, (short)0, acc[nt], false, false);
      acc[nt] = __builtin_amdgcn_wmma_f32_16x16x32_bf16(
          false, ahi, false, bhi, (short)0, acc[nt], false, false);
    }
    __syncthreads();
  }
  gemm_epilogue<MODE>(acc, bias, C, M, N, m0, n0, wave, lane, pos_buf, norm_w);
}

// ---------------------------------------------------------------------------
// 5) exact-fp32 WMMA GEMM (V_WMMA_F32_16X16X4_F32) for the small KV GEMM
// ---------------------------------------------------------------------------
template<int MODE>
__global__ __launch_bounds__(256) void gemm_f32_kernel(
    const float* __restrict__ A, const float* __restrict__ B,
    const float* __restrict__ bias, float* __restrict__ C,
    int M, int N, int K,
    const float* __restrict__ pos_buf, const float* __restrict__ norm_w)
{
  __shared__ float ldsA[BM * LDA_P];
  __shared__ float ldsB[BK * LDB_P];

  const int tid  = threadIdx.x;
  const int wave = tid >> 5;
  const int lane = tid & 31;
  const int lhalf = lane >> 4, lmod = lane & 15;
  const int n0 = blockIdx.x * BN;
  const int m0 = blockIdx.y * BM;

  v8f acc[8] = {};

  const int ar = tid >> 1, ac = (tid & 1) * 16;
  const int br = tid >> 3, bc = (tid & 7) * 16;

  for (int kb = 0; kb < K; kb += BK) {
    {
      float4* dst = (float4*)(ldsA + ar * LDA_P + ac);
      int grow = m0 + ar;
      if (grow < M) {
        const float4* src = (const float4*)(A + (size_t)grow * K + kb + ac);
        #pragma unroll
        for (int i = 0; i < 4; i++) dst[i] = src[i];
      } else {
        float4 z = make_float4(0.f, 0.f, 0.f, 0.f);
        #pragma unroll
        for (int i = 0; i < 4; i++) dst[i] = z;
      }
    }
    {
      const float4* src = (const float4*)(B + (size_t)(kb + br) * N + n0 + bc);
      float4* dst = (float4*)(ldsB + br * LDB_P + bc);
      #pragma unroll
      for (int i = 0; i < 4; i++) dst[i] = src[i];
    }
    __syncthreads();
    const int arow = wave * 16 + lmod;
    #pragma unroll
    for (int ks = 0; ks < BK; ks += 4) {
      v2f afrag = *(const v2f*)(ldsA + arow * LDA_P + ks + 2 * lhalf);
      #pragma unroll
      for (int nt = 0; nt < 8; nt++) {
        int col = nt * 16 + lmod;
        v2f bfrag;
        bfrag.x = ldsB[(ks + 2 * lhalf) * LDB_P + col];
        bfrag.y = ldsB[(ks + 2 * lhalf + 1) * LDB_P + col];
        acc[nt] = __builtin_amdgcn_wmma_f32_16x16x4_f32(
            false, afrag, false, bfrag, (short)0, acc[nt], false, false);
      }
    }
    __syncthreads();
  }
  gemm_epilogue<MODE>(acc, bias, C, M, N, m0, n0, wave, lane, pos_buf, norm_w);
}

// ---------------------------------------------------------------------------
// 6) attention: per (frame, head, 256-token chunk); K/V in LDS, in-place out
// ---------------------------------------------------------------------------
__global__ __launch_bounds__(256) void attn_kernel(float* __restrict__ q,
                                                   const float* __restrict__ kv)
{
  __shared__ float kl[32 * 128];
  __shared__ float vl[32 * 128];
  __shared__ float ql[256 * 20];

  const int t = threadIdx.x;
  const int f = blockIdx.z, h = blockIdx.y;
  const int s0 = blockIdx.x * 256;
  const int Cd = 3072, KVN = 6144, S = 1024;

  { int row = t >> 3, cg = (t & 7) * 16;
    const float4* ks = (const float4*)(kv + (size_t)(f * 32 + row) * KVN + h * 128 + cg);
    const float4* vs = (const float4*)(kv + (size_t)(f * 32 + row) * KVN + Cd + h * 128 + cg);
    float4* kd = (float4*)(kl + row * 128 + cg);
    float4* vd = (float4*)(vl + row * 128 + cg);
    #pragma unroll
    for (int i = 0; i < 4; i++) { kd[i] = ks[i]; vd[i] = vs[i]; } }
  float sc[32];
  #pragma unroll
  for (int j = 0; j < 32; j++) sc[j] = 0.f;
  __syncthreads();

  for (int c = 0; c < 8; c++) {
    #pragma unroll
    for (int i = 0; i < 4; i++) {
      int f4 = t + 256 * i;
      int row = f4 >> 2, cc = (f4 & 3) * 4;
      float4 vq = *(const float4*)(q + (size_t)(f * S + s0 + row) * Cd + h * 128 + c * 16 + cc);
      *(float4*)(ql + row * 20 + cc) = vq;
    }
    __syncthreads();
    float qr[16];
    #pragma unroll
    for (int i = 0; i < 4; i++) *(float4*)(qr + 4 * i) = *(const float4*)(ql + t * 20 + 4 * i);
    #pragma unroll
    for (int j = 0; j < 32; j++) {
      float a = 0.f;
      #pragma unroll
      for (int d2 = 0; d2 < 16; d2 += 4) {
        float4 kk = *(const float4*)(kl + j * 128 + c * 16 + d2);
        a += qr[d2] * kk.x + qr[d2+1] * kk.y + qr[d2+2] * kk.z + qr[d2+3] * kk.w;
      }
      sc[j] += a;
    }
    __syncthreads();
  }

  const float scale = 0.08838834764831845f;
  float m = sc[0];
  #pragma unroll
  for (int j = 1; j < 32; j++) m = fmaxf(m, sc[j]);
  float sum = 0.f;
  #pragma unroll
  for (int j = 0; j < 32; j++) { sc[j] = __expf((sc[j] - m) * scale); sum += sc[j]; }
  float inv = 1.0f / sum;
  #pragma unroll
  for (int j = 0; j < 32; j++) sc[j] *= inv;

  float* orow = q + (size_t)(f * S + s0 + t) * Cd + h * 128;
  #pragma unroll
  for (int dg = 0; dg < 128; dg += 4) {
    float4 a = make_float4(0.f, 0.f, 0.f, 0.f);
    #pragma unroll
    for (int j = 0; j < 32; j++) {
      float4 vv = *(const float4*)(vl + j * 128 + dg);
      a.x += sc[j] * vv.x; a.y += sc[j] * vv.y; a.z += sc[j] * vv.z; a.w += sc[j] * vv.w;
    }
    *(float4*)(orow + dg) = a;
  }
}

// ---------------------------------------------------------------------------
extern "C" void kernel_launch(void* const* d_in, const int* in_sizes, int n_in,
                              void* d_out, int out_size, void* d_ws, size_t ws_size,
                              hipStream_t stream) {
  (void)in_sizes; (void)n_in; (void)out_size; (void)ws_size;
  const float* x      = (const float*)d_in[0];
  const float* cond   = (const float*)d_in[1];
  const float* xmap   = (const float*)d_in[2];
  const float* q_w    = (const float*)d_in[3];
  const float* q_b    = (const float*)d_in[4];
  const float* q_norm = (const float*)d_in[5];
  const float* kv_w   = (const float*)d_in[6];
  const float* kv_b   = (const float*)d_in[7];
  const float* k_norm = (const float*)d_in[8];
  const float* proj_w = (const float*)d_in[9];
  const float* proj_b = (const float*)d_in[10];

  const int Ntok = 21 * 1024, Cdim = 3072;

  char* ws = (char*)d_ws;
  float*  mm4   = (float*)ws;                                     // 16 B
  float*  pos   = (float*)(ws + 256);                             // 86016 B
  float*  qbuf  = (float*)(ws + 86272);                           // 21504*3072 f32
  float*  kvbuf = (float*)(ws + 86272 + (size_t)21504*3072*4);    // 672*6144 f32
  char*   wsp   = ws + 86272 + (size_t)21504*3072*4 + (size_t)672*6144*4;
  __bf16* qwHi  = (__bf16*)(wsp);                                 // 3072*3072 bf16 each
  __bf16* qwLo  = (__bf16*)(wsp + (size_t)3072*3072*2);
  __bf16* pwHi  = (__bf16*)(wsp + (size_t)3072*3072*4);
  __bf16* pwLo  = (__bf16*)(wsp + (size_t)3072*3072*6);

  minmax_kernel<<<1, 256, 0, stream>>>(xmap, Ntok, mm4);
  pos_kernel<<<(Ntok + 255) / 256, 256, 0, stream>>>(xmap, Ntok, mm4, pos);

  dim3 gt(Cdim / 32, Cdim / 32);                  // 96 x 96
  transpose_split_kernel<<<gt, 256, 0, stream>>>(q_w,    qwHi, qwLo, Cdim, Cdim);
  transpose_split_kernel<<<gt, 256, 0, stream>>>(proj_w, pwHi, pwLo, Cdim, Cdim);

  dim3 gq(Cdim / BN, Ntok / BM);                  // 24 x 168
  gemm_bf16x3_kernel<0><<<gq, 256, 0, stream>>>(x, qwHi, qwLo, q_b, qbuf,
                                                Ntok, Cdim, Cdim, pos, q_norm);

  dim3 gkv(6144 / BN, (672 + BM - 1) / BM);       // 48 x 6
  gemm_f32_kernel<1><<<gkv, 256, 0, stream>>>(cond, kv_w, kv_b, kvbuf,
                                              672, 6144, 768, nullptr, k_norm);

  dim3 ga(4, 24, 21);                             // (s-chunk, head, frame)
  attn_kernel<<<ga, 256, 0, stream>>>(qbuf, kvbuf);

  gemm_bf16x3_kernel<2><<<gq, 256, 0, stream>>>(qbuf, pwHi, pwLo, proj_b, (float*)d_out,
                                                Ntok, Cdim, Cdim, nullptr, nullptr);
}
